// SSMLayer_46145128628525
// MI455X (gfx1250) — compile-verified, matrix operand order checked by reading, not run on hardware
//
#include <hip/hip_runtime.h>
#include <stdint.h>

// ---------------- problem constants ----------------
#define BATCH   8
#define SEQLEN  8192
#define DMODEL  512
#define DSTATE  64
#define MTOT    (BATCH * SEQLEN)      // 65536 rows
#define CHUNK   128
#define NCHUNK  (SEQLEN / CHUNK)      // 64
#define LN_EPS  1e-3f

// ---------------- WMMA types ----------------
typedef __bf16 bf16_t;
typedef bf16_t v16bf __attribute__((ext_vector_type(16)));
typedef float  v8f   __attribute__((ext_vector_type(8)));

union FragU { v16bf v; uint4 q[2]; };

// round-to-nearest-even f32 -> bf16 bits
__device__ __forceinline__ uint16_t f2bf(float f) {
    uint32_t u = __float_as_uint(f);
    uint32_t r = u + 0x7FFFu + ((u >> 16) & 1u);
    return (uint16_t)(r >> 16);
}

__device__ __forceinline__ float softplusf(float x) {
    return (x > 20.f) ? x : log1pf(expf(x));
}

// ---------------- weight convert + transpose: (K,N) f32 -> (N,K) bf16 ----------------
__global__ void wcvt_kernel(const float* __restrict__ w, uint16_t* __restrict__ wt,
                            int K, int N) {
    int idx = blockIdx.x * blockDim.x + threadIdx.x;
    if (idx >= K * N) return;
    int k = idx / N, n = idx % N;
    wt[(size_t)n * K + k] = f2bf(w[idx]);
}

// ---------------- LayerNorm: one wave per row of 512, output bf16 ----------------
__global__ __launch_bounds__(256) void ln_kernel(const float* __restrict__ x,
                                                 const float* __restrict__ gamma,
                                                 const float* __restrict__ beta,
                                                 uint16_t* __restrict__ out) {
    int lane = threadIdx.x & 31;
    int row  = blockIdx.x * 8 + (threadIdx.x >> 5);
    const float* base = x + (size_t)row * DMODEL;
    float r[16];
    float s = 0.f, sq = 0.f;
#pragma unroll
    for (int i = 0; i < 16; ++i) {
        float v = base[lane + 32 * i];
        r[i] = v; s += v; sq += v * v;
    }
#pragma unroll
    for (int o = 16; o > 0; o >>= 1) {
        s  += __shfl_xor(s, o, 32);
        sq += __shfl_xor(sq, o, 32);
    }
    float mean = s * (1.f / DMODEL);
    float var  = sq * (1.f / DMODEL) - mean * mean;
    float rstd = rsqrtf(var + LN_EPS);
    uint16_t* ob = out + (size_t)row * DMODEL;
#pragma unroll
    for (int i = 0; i < 16; ++i) {
        int col = lane + 32 * i;
        float xn = (r[i] - mean) * rstd * gamma[col] + beta[col];
        ob[col] = f2bf(xn);
    }
}

// ---------------- WMMA GEMM: C[M,N] = A[M,K](bf16) * Bt[N,K](bf16) + epilogue ----
// K and LDC are compile-time (powers of two) so all address math strength-reduces.
// MODE 0: v += bias[n];                 store f32 + bf16     (z = xn@W_in + b_in)
// MODE 1:                               store f32            (x_state = z@W_xs)
// MODE 2: v += Dvec[n]*aux[m*ldc+n];    store bf16           (ymid = yC@W_so + D*z)
// MODE 3: v += bias[n] + aux[m*ldc+n];  store f32            (out = ymid@W_out + b + res)
template <int MODE, int BMW, int BNW, int WMT, int WNT, int K, int LDC>
__global__ __launch_bounds__(BMW * BNW * 32)
void gemm_bf16_wmma(const uint16_t* __restrict__ Abuf,
                    const uint16_t* __restrict__ Bbuf,
                    const float* __restrict__ bias,
                    const float* __restrict__ aux,
                    const float* __restrict__ Dvec,
                    float* __restrict__ outF,
                    uint16_t* __restrict__ outB) {
    const int lane  = threadIdx.x & 31;
    const int wave  = threadIdx.x >> 5;
    const int wm    = wave % BMW;
    const int wn    = wave / BMW;
    const int waveM = blockIdx.y * (BMW * WMT * 16) + wm * (WMT * 16);
    const int waveN = blockIdx.x * (BNW * WNT * 16) + wn * (WNT * 16);

    const int rowA  = lane & 15;        // M (or N for B-frags) within 16
    const int khalf = (lane >> 4) * 8;  // which K-half this lane holds first

    v8f acc[WMT][WNT];
#pragma unroll
    for (int i = 0; i < WMT; ++i)
#pragma unroll
        for (int j = 0; j < WNT; ++j)
#pragma unroll
            for (int e = 0; e < 8; ++e) acc[i][j][e] = 0.f;

    const uint16_t* Abase = Abuf + (size_t)(waveM + rowA) * K + khalf;
    const uint16_t* Bbase = Bbuf + (size_t)(waveN + rowA) * K + khalf;

    for (int k0 = 0; k0 < K; k0 += 32) {
        v16bf af[WMT], bfm[WNT];
#pragma unroll
        for (int i = 0; i < WMT; ++i) {
            const uint16_t* p = Abase + (size_t)i * 16 * K + k0;
            FragU u;
            u.q[0] = *reinterpret_cast<const uint4*>(p);
            u.q[1] = *reinterpret_cast<const uint4*>(p + 16);
            af[i] = u.v;
        }
#pragma unroll
        for (int j = 0; j < WNT; ++j) {
            const uint16_t* p = Bbase + (size_t)j * 16 * K + k0;
            FragU u;
            u.q[0] = *reinterpret_cast<const uint4*>(p);
            u.q[1] = *reinterpret_cast<const uint4*>(p + 16);
            bfm[j] = u.v;
        }
#pragma unroll
        for (int i = 0; i < WMT; ++i)
#pragma unroll
            for (int j = 0; j < WNT; ++j)
                acc[i][j] = __builtin_amdgcn_wmma_f32_16x16x32_bf16(
                    false, af[i], false, bfm[j], (short)0, acc[i][j], false, false);
    }

    // epilogue: VGPR e holds M = e + 8*(lane>=16), N = lane&15 within each 16x16 tile
    const int moff = (lane >> 4) * 8;
    const int ncol = lane & 15;
#pragma unroll
    for (int i = 0; i < WMT; ++i) {
#pragma unroll
        for (int j = 0; j < WNT; ++j) {
            int gn = waveN + j * 16 + ncol;
#pragma unroll
            for (int e = 0; e < 8; ++e) {
                int gm = waveM + i * 16 + moff + e;
                size_t idx = (size_t)gm * LDC + gn;
                float v = acc[i][j][e];
                if (MODE == 0) {
                    v += bias[gn];
                    outF[idx] = v;
                    outB[idx] = f2bf(v);
                } else if (MODE == 1) {
                    outF[idx] = v;
                } else if (MODE == 2) {
                    v += Dvec[gn] * aux[idx];
                    outB[idx] = f2bf(v);
                } else { // MODE == 3
                    v += bias[gn] + aux[idx];
                    outF[idx] = v;
                }
            }
        }
    }
}

// ---------------- SSM discretization helper ----------------
__device__ __forceinline__ void ssm_params(const float* A_log, const float* log_delta,
                                           const float* Bv, int n,
                                           float& Abar, float& Bbar) {
    float A     = -expf(A_log[n]);
    float delta = softplusf(log_delta[n]);
    float dA2   = 0.5f * delta * A;
    float denom = 1.f - dA2;
    Abar = (1.f + dA2) / denom;
    Bbar = delta / denom * Bv[n];
}

// ---------------- scan pass 1: chunk-local recurrence, in-place; emit chunk carry ----
__global__ __launch_bounds__(DSTATE) void scan1_kernel(float* __restrict__ xs,
                                                       float* __restrict__ carry,
                                                       const float* __restrict__ A_log,
                                                       const float* __restrict__ log_delta,
                                                       const float* __restrict__ Bv) {
    int b = blockIdx.x / NCHUNK;
    int c = blockIdx.x % NCHUNK;
    int n = threadIdx.x;
    float Abar, Bbar;
    ssm_params(A_log, log_delta, Bv, n, Abar, Bbar);
    float h = 0.f;
    float* p = xs + ((size_t)b * SEQLEN + (size_t)c * CHUNK) * DSTATE + n;
    for (int t = 0; t < CHUNK; ++t) {
        float u = *p;
        h = Abar * h + Bbar * u;
        *p = h;
        p += DSTATE;
    }
    carry[((size_t)b * NCHUNK + c) * DSTATE + n] = h;
}

// ---------------- scan pass 2: sequential carry combine across chunks ----------------
__global__ __launch_bounds__(BATCH * DSTATE) void scan2_kernel(const float* __restrict__ carry,
                                                               float* __restrict__ carry_in,
                                                               const float* __restrict__ A_log,
                                                               const float* __restrict__ log_delta,
                                                               const float* __restrict__ Bv) {
    int tid = threadIdx.x;
    int b = tid >> 6, n = tid & 63;
    float Abar, Bbar;
    ssm_params(A_log, log_delta, Bv, n, Abar, Bbar);
    float Ak = powf(Abar, (float)CHUNK);
    float cin = 0.f;
    for (int c = 0; c < NCHUNK; ++c) {
        size_t idx = ((size_t)b * NCHUNK + c) * DSTATE + n;
        carry_in[idx] = cin;
        cin = Ak * cin + carry[idx];
    }
}

// ---------------- scan pass 3: apply carry correction, scale by C, emit bf16 ----------------
__global__ __launch_bounds__(DSTATE) void scan3_kernel(const float* __restrict__ xs,
                                                       const float* __restrict__ carry_in,
                                                       const float* __restrict__ A_log,
                                                       const float* __restrict__ log_delta,
                                                       const float* __restrict__ Bv,
                                                       const float* __restrict__ Cv,
                                                       uint16_t* __restrict__ out) {
    int b = blockIdx.x / NCHUNK;
    int c = blockIdx.x % NCHUNK;
    int n = threadIdx.x;
    float Abar, Bbar;
    ssm_params(A_log, log_delta, Bv, n, Abar, Bbar);
    float cin = carry_in[((size_t)b * NCHUNK + c) * DSTATE + n];
    float Cn  = Cv[n];
    float Ap  = Abar;
    size_t base = ((size_t)b * SEQLEN + (size_t)c * CHUNK) * DSTATE + n;
    for (int t = 0; t < CHUNK; ++t) {
        float h = xs[base] + Ap * cin;
        out[base] = f2bf(h * Cn);
        Ap *= Abar;
        base += DSTATE;
    }
}

// ---------------- host-side launch ----------------
extern "C" void kernel_launch(void* const* d_in, const int* in_sizes, int n_in,
                              void* d_out, int out_size, void* d_ws, size_t ws_size,
                              hipStream_t stream) {
    const float* x         = (const float*)d_in[0];
    const float* ln_gamma  = (const float*)d_in[1];
    const float* ln_beta   = (const float*)d_in[2];
    const float* W_in      = (const float*)d_in[3];
    const float* b_in      = (const float*)d_in[4];
    const float* W_xs      = (const float*)d_in[5];
    const float* A_log     = (const float*)d_in[6];
    const float* log_delta = (const float*)d_in[7];
    const float* Bv        = (const float*)d_in[8];
    const float* Cv        = (const float*)d_in[9];
    const float* Dv        = (const float*)d_in[10];
    const float* W_so      = (const float*)d_in[11];
    const float* W_out     = (const float*)d_in[12];
    const float* b_out     = (const float*)d_in[13];
    float* out = (float*)d_out;

    // ---- workspace layout (bytes) ----
    char* ws = (char*)d_ws;
    size_t off = 0;
    auto take = [&](size_t bytes) { size_t o = off; off += (bytes + 255) & ~(size_t)255; return o; };
    uint16_t* bufA    = (uint16_t*)(ws + take((size_t)MTOT * DMODEL * 2)); // xn, later ymid
    float*    zf      = (float*)   (ws + take((size_t)MTOT * DMODEL * 4));
    uint16_t* zb      = (uint16_t*)(ws + take((size_t)MTOT * DMODEL * 2));
    float*    xsbuf   = (float*)   (ws + take((size_t)MTOT * DSTATE * 4));
    uint16_t* yssm    = (uint16_t*)(ws + take((size_t)MTOT * DSTATE * 2));
    uint16_t* WinT    = (uint16_t*)(ws + take((size_t)DMODEL * DMODEL * 2));
    uint16_t* WxsT    = (uint16_t*)(ws + take((size_t)DMODEL * DSTATE * 2));
    uint16_t* WsoT    = (uint16_t*)(ws + take((size_t)DSTATE * DMODEL * 2));
    uint16_t* WoutT   = (uint16_t*)(ws + take((size_t)DMODEL * DMODEL * 2));
    float*    carry   = (float*)   (ws + take((size_t)BATCH * NCHUNK * DSTATE * 4));
    float*    carryIn = (float*)   (ws + take((size_t)BATCH * NCHUNK * DSTATE * 4));

    // ---- 1. weight convert + transpose to bf16 (N,K) ----
    wcvt_kernel<<<(DMODEL * DMODEL + 255) / 256, 256, 0, stream>>>(W_in,  WinT,  DMODEL, DMODEL);
    wcvt_kernel<<<(DMODEL * DSTATE + 255) / 256, 256, 0, stream>>>(W_xs,  WxsT,  DMODEL, DSTATE);
    wcvt_kernel<<<(DSTATE * DMODEL + 255) / 256, 256, 0, stream>>>(W_so,  WsoT,  DSTATE, DMODEL);
    wcvt_kernel<<<(DMODEL * DMODEL + 255) / 256, 256, 0, stream>>>(W_out, WoutT, DMODEL, DMODEL);

    // ---- 2. LayerNorm -> bf16 ----
    ln_kernel<<<MTOT / 8, 256, 0, stream>>>(x, ln_gamma, ln_beta, bufA);

    // ---- 3. z = xn @ W_in + b_in  (f32 + bf16 copies) ----
    {
        dim3 grid(DMODEL / 128, MTOT / 128);
        gemm_bf16_wmma<0, 4, 2, 2, 4, DMODEL, DMODEL><<<grid, 256, 0, stream>>>(
            bufA, WinT, b_in, nullptr, nullptr, zf, zb);
    }
    // ---- 4. x_state = z @ W_xs  (N=64) ----
    {
        dim3 grid(1, MTOT / 256);
        gemm_bf16_wmma<1, 8, 1, 2, 4, DMODEL, DSTATE><<<grid, 256, 0, stream>>>(
            zb, WxsT, nullptr, nullptr, nullptr, xsbuf, nullptr);
    }
    // ---- 5. chunked diagonal SSM scan ----
    scan1_kernel<<<BATCH * NCHUNK, DSTATE, 0, stream>>>(xsbuf, carry, A_log, log_delta, Bv);
    scan2_kernel<<<1, BATCH * DSTATE, 0, stream>>>(carry, carryIn, A_log, log_delta, Bv);
    scan3_kernel<<<BATCH * NCHUNK, DSTATE, 0, stream>>>(xsbuf, carryIn, A_log, log_delta, Bv, Cv, yssm);

    // ---- 6. ymid = (y_state*C) @ W_so + D*z  -> bf16 (reuse bufA) ----
    {
        dim3 grid(DMODEL / 128, MTOT / 128);
        gemm_bf16_wmma<2, 4, 2, 2, 4, DSTATE, DMODEL><<<grid, 256, 0, stream>>>(
            yssm, WsoT, nullptr, zf, Dv, nullptr, bufA);
    }
    // ---- 7. out = ymid @ W_out + b_out + residual ----
    {
        dim3 grid(DMODEL / 128, MTOT / 128);
        gemm_bf16_wmma<3, 4, 2, 2, 4, DMODEL, DMODEL><<<grid, 256, 0, stream>>>(
            bufA, WoutT, b_out, x, nullptr, out, nullptr);
    }
}
